// SelfAttention_86638080294969
// MI455X (gfx1250) — matrix operand order repaired and checked by reading, not compiled, not run on hardware
//
#include <hip/hip_runtime.h>

typedef __bf16 bf16_t;
typedef __bf16 bf16x8_t  __attribute__((ext_vector_type(8)));
typedef __bf16 bf16x16_t __attribute__((ext_vector_type(16)));
typedef float  f32x8_t   __attribute__((ext_vector_type(8)));

#define TILE_M 128
#define TILE_N 128
#define TILE_K 32
#define LDS_STRIDE 40   // 32 + 8 halves padding (80B rows; conflict-free frag reads)

// ---------------------------------------------------------------------------
// Block reductions (256 threads)
// ---------------------------------------------------------------------------
__device__ __forceinline__ float block_reduce_sum(float v, float* sm, int tid) {
    sm[tid] = v; __syncthreads();
    #pragma unroll
    for (int s = 128; s > 0; s >>= 1) {
        if (tid < s) sm[tid] += sm[tid + s];
        __syncthreads();
    }
    float r = sm[0]; __syncthreads();
    return r;
}
__device__ __forceinline__ float block_reduce_max(float v, float* sm, int tid) {
    sm[tid] = v; __syncthreads();
    #pragma unroll
    for (int s = 128; s > 0; s >>= 1) {
        if (tid < s) sm[tid] = fmaxf(sm[tid], sm[tid + s]);
        __syncthreads();
    }
    float r = sm[0]; __syncthreads();
    return r;
}

// ---------------------------------------------------------------------------
// LayerNorm (one block per row of D=1024) -> bf16
// ---------------------------------------------------------------------------
__global__ __launch_bounds__(256) void ln_bf16_kernel(
    const float* __restrict__ x, const float* __restrict__ g,
    const float* __restrict__ b, bf16_t* __restrict__ xn, int D)
{
    __shared__ float sm[256];
    const int row = blockIdx.x, tid = threadIdx.x;
    const float* xr = x + (size_t)row * D;
    float4 v = *(const float4*)(xr + tid * 4);

    float s = block_reduce_sum(v.x + v.y + v.z + v.w, sm, tid);
    const float mean = s / (float)D;
    const float dx = v.x - mean, dy = v.y - mean, dz = v.z - mean, dw = v.w - mean;
    float sq = block_reduce_sum(dx*dx + dy*dy + dz*dz + dw*dw, sm, tid);
    const float rstd = rsqrtf(sq / (float)D + 1e-5f);

    float4 gv = *(const float4*)(g + tid * 4);
    float4 bv = *(const float4*)(b + tid * 4);
    bf16_t* o = xn + (size_t)row * D + tid * 4;
    o[0] = (bf16_t)(dx * rstd * gv.x + bv.x);
    o[1] = (bf16_t)(dy * rstd * gv.y + bv.y);
    o[2] = (bf16_t)(dz * rstd * gv.z + bv.z);
    o[3] = (bf16_t)(dw * rstd * gv.w + bv.w);
}

// ---------------------------------------------------------------------------
// fp32 -> bf16 cast
// ---------------------------------------------------------------------------
__global__ __launch_bounds__(256) void cast_bf16_kernel(
    const float* __restrict__ src, bf16_t* __restrict__ dst, int n)
{
    int i = (blockIdx.x * 256 + threadIdx.x) * 4;
    if (i + 3 < n) {
        float4 v = *(const float4*)(src + i);
        dst[i + 0] = (bf16_t)v.x; dst[i + 1] = (bf16_t)v.y;
        dst[i + 2] = (bf16_t)v.z; dst[i + 3] = (bf16_t)v.w;
    }
}

// ---------------------------------------------------------------------------
// Row softmax over f32 scores (S=2048), write bf16 P IN PLACE (row stride 2*S
// halves == the f32 row's byte footprint). One block per row; all reads finish
// before the reduction barriers, stores only after, so aliasing is safe.
// ---------------------------------------------------------------------------
__global__ __launch_bounds__(256) void softmax_inplace_kernel(
    float* __restrict__ scores, int S)
{
    __shared__ float sm[256];
    const int row = blockIdx.x, tid = threadIdx.x;
    float* sr = scores + (size_t)row * S;
    float4 v0 = *(const float4*)(sr + tid * 4);
    float4 v1 = *(const float4*)(sr + (S / 2) + tid * 4);

    float m = fmaxf(fmaxf(fmaxf(v0.x, v0.y), fmaxf(v0.z, v0.w)),
                    fmaxf(fmaxf(v1.x, v1.y), fmaxf(v1.z, v1.w)));
    m = block_reduce_max(m, sm, tid);

    float e[8];
    e[0]=__expf(v0.x-m); e[1]=__expf(v0.y-m); e[2]=__expf(v0.z-m); e[3]=__expf(v0.w-m);
    e[4]=__expf(v1.x-m); e[5]=__expf(v1.y-m); e[6]=__expf(v1.z-m); e[7]=__expf(v1.w-m);
    float sum = block_reduce_sum(e[0]+e[1]+e[2]+e[3]+e[4]+e[5]+e[6]+e[7], sm, tid);
    const float inv = 1.0f / sum;

    bf16_t* pr = (bf16_t*)sr;   // in-place bf16 row, logical stride 2*S halves
    #pragma unroll
    for (int k = 0; k < 4; ++k)  pr[tid * 4 + k]           = (bf16_t)(e[k] * inv);
    #pragma unroll
    for (int k = 0; k < 4; ++k)  pr[(S / 2) + tid * 4 + k] = (bf16_t)(e[4 + k] * inv);
}

// ---------------------------------------------------------------------------
// Batched bf16 GEMM-NT via v_wmma_f32_16x16x32_bf16:
//   C[m][n] = alpha * sum_k A[m][k]*Bt[n][k] (+ bias[n])
// A: M x K row-major, Bt: N x K row-major (B column data contiguous).
// OUT_MODE: 0 = f32 row-major, 1 = bf16 row-major, 2 = bf16 transposed C[n][m].
// Workgroup tile 128x128 (8 waves, 4M x 2N), wave tile 32x64 (2x4 frags =
// 8 WMMA per K-step, A frags reused across 4 B frags), double-buffered LDS.
// ---------------------------------------------------------------------------
template<int OUT_MODE, bool HAS_BIAS>
__global__ __launch_bounds__(256) void gemm_bf16_nt(
    const bf16_t* __restrict__ A, const bf16_t* __restrict__ Bt,
    void* __restrict__ Cout, const float* __restrict__ bias,
    int M, int N, int K, int lda, int ldb, int ldc,
    long batchA, long batchB, long batchC, float alpha)
{
    __shared__ bf16_t As[2][TILE_M * LDS_STRIDE];   // 2 x 10 KB
    __shared__ bf16_t Bs[2][TILE_N * LDS_STRIDE];   // 2 x 10 KB

    const int tid   = threadIdx.x;
    const int lane  = tid & 31;
    const int wid   = tid >> 5;
    const int waveM = (wid & 3) * 32;   // 4 waves along M: 0,32,64,96
    const int waveN = (wid >> 2) * 64;  // 2 waves along N: 0,64
    const int nBase = blockIdx.x * TILE_N;
    const int mBase = blockIdx.y * TILE_M;
    const int bz    = blockIdx.z;

    A  += (size_t)bz * batchA;
    Bt += (size_t)bz * batchB;

    // Global->LDS staging map: 16B chunks of 8 halves; 128x32 halves per matrix
    // = 512 chunks = 2 chunks/thread for each of A and B.
    const int r0   = tid >> 2;              // 0..63
    const int r1   = r0 + 64;               // 64..127
    const int ccol = (tid & 3) * 8;         // 0,8,16,24

    bf16x8_t stA0, stA1, stB0, stB1;
    auto loadG = [&](int kt) {
        const size_t kb = (size_t)kt * TILE_K;
        stA0 = *(const bf16x8_t*)(A  + (size_t)(mBase + r0) * lda + kb + ccol);
        stA1 = *(const bf16x8_t*)(A  + (size_t)(mBase + r1) * lda + kb + ccol);
        stB0 = *(const bf16x8_t*)(Bt + (size_t)(nBase + r0) * ldb + kb + ccol);
        stB1 = *(const bf16x8_t*)(Bt + (size_t)(nBase + r1) * ldb + kb + ccol);
    };
    auto storeL = [&](int buf) {
        *(bf16x8_t*)&As[buf][r0 * LDS_STRIDE + ccol] = stA0;
        *(bf16x8_t*)&As[buf][r1 * LDS_STRIDE + ccol] = stA1;
        *(bf16x8_t*)&Bs[buf][r0 * LDS_STRIDE + ccol] = stB0;
        *(bf16x8_t*)&Bs[buf][r1 * LDS_STRIDE + ccol] = stB1;
    };

    f32x8_t acc[2][4] = {};
    const int nk = K / TILE_K;

    loadG(0);
    storeL(0);
    for (int kt = 0; kt < nk; ++kt) {
        __syncthreads();                       // tile kt visible in LDS[buf]
        const int buf = kt & 1;
        if (kt + 1 < nk) loadG(kt + 1);        // overlap global fetch w/ math

        // A fragments (16x32 bf16): lane m = lane&15; halves 0..7 = K[cb..cb+7],
        // halves 8..15 = K[cb+16..cb+23], cb = (lane>=16)*8.
        bf16x16_t af[2], bfr[4];
        #pragma unroll
        for (int i = 0; i < 2; ++i) {
            const bf16_t* p = &As[buf][(waveM + i * 16 + (lane & 15)) * LDS_STRIDE
                                       + ((lane >> 4) * 8)];
            bf16x8_t lo = *(const bf16x8_t*)p;
            bf16x8_t hi = *(const bf16x8_t*)(p + 16);
            #pragma unroll
            for (int e = 0; e < 8; ++e) { af[i][e] = lo[e]; af[i][e + 8] = hi[e]; }
        }
        // B fragments (32x16 bf16): lane n = lane&15; halves 0..7 = K[cb..cb+7],
        // halves 8..15 = K[cb+16..cb+23], cb = (lane>=16)*8.
        #pragma unroll
        for (int j = 0; j < 4; ++j) {
            const bf16_t* p = &Bs[buf][(waveN + j * 16 + (lane & 15)) * LDS_STRIDE
                                       + ((lane >> 4) * 8)];
            bf16x8_t lo = *(const bf16x8_t*)p;
            bf16x8_t hi = *(const bf16x8_t*)(p + 16);
            #pragma unroll
            for (int e = 0; e < 8; ++e) { bfr[j][e] = lo[e]; bfr[j][e + 8] = hi[e]; }
        }
        #pragma unroll
        for (int i = 0; i < 2; ++i)
            #pragma unroll
            for (int j = 0; j < 4; ++j)
                acc[i][j] = __builtin_amdgcn_wmma_f32_16x16x32_bf16(
                    false, af[i], false, bfr[j], (short)0, acc[i][j], false, false);

        if (kt + 1 < nk) storeL(buf ^ 1);      // other buffer; next barrier fences
    }

    // C layout: VGPR r -> m = r + 8*(lane>=16); n = lane&15.
    #pragma unroll
    for (int i = 0; i < 2; ++i) {
        #pragma unroll
        for (int j = 0; j < 4; ++j) {
            const int n = nBase + waveN + j * 16 + (lane & 15);
            const float bterm = HAS_BIAS ? bias[n] : 0.0f;
            #pragma unroll
            for (int r = 0; r < 8; ++r) {
                const int m = mBase + waveM + i * 16 + ((lane >> 4) << 3) + r;
                const float v = acc[i][j][r] * alpha + bterm;
                if (OUT_MODE == 0)
                    ((float*)Cout)[(size_t)bz * batchC + (size_t)m * ldc + n] = v;
                else if (OUT_MODE == 1)
                    ((bf16_t*)Cout)[(size_t)bz * batchC + (size_t)m * ldc + n] = (bf16_t)v;
                else
                    ((bf16_t*)Cout)[(size_t)bz * batchC + (size_t)n * ldc + m] = (bf16_t)v;
            }
        }
    }
}

// ---------------------------------------------------------------------------
// Host-side orchestration
// ---------------------------------------------------------------------------
extern "C" void kernel_launch(void* const* d_in, const int* in_sizes, int n_in,
                              void* d_out, int out_size, void* d_ws, size_t ws_size,
                              hipStream_t stream) {
    const float* x     = (const float*)d_in[0];
    const float* gamma = (const float*)d_in[1];
    const float* beta  = (const float*)d_in[2];
    const float* Wq    = (const float*)d_in[3];
    const float* bq    = (const float*)d_in[4];
    const float* Wk    = (const float*)d_in[5];
    const float* bk    = (const float*)d_in[6];
    const float* Wv    = (const float*)d_in[7];
    const float* bv    = (const float*)d_in[8];
    const float* Wo    = (const float*)d_in[9];
    const float* bo    = (const float*)d_in[10];
    float* out = (float*)d_out;

    const int D = 1024, B = 4, S = 2048;
    const int rows = B * S;                     // 8192
    const float inv_sqrt_d = 0.03125f;          // 1/sqrt(1024)

    char* ws = (char*)d_ws;
    bf16_t* xn  = (bf16_t*)(ws);                         // 16 MB  xn bf16
    bf16_t* Wqb = (bf16_t*)(ws + (16ull << 20));         //  2 MB
    bf16_t* Wkb = (bf16_t*)(ws + (18ull << 20));         //  2 MB
    bf16_t* Wvb = (bf16_t*)(ws + (20ull << 20));         //  2 MB
    bf16_t* Wob = (bf16_t*)(ws + (22ull << 20));         //  2 MB
    bf16_t* Qb  = (bf16_t*)(ws + (24ull << 20));         // 16 MB  (reused as attn-out)
    bf16_t* Kb  = (bf16_t*)(ws + (40ull << 20));         // 16 MB
    bf16_t* Vt  = (bf16_t*)(ws + (56ull << 20));         // 16 MB  V^T: [D][rows]
    float*  Sc  = (float*) (ws + (72ull << 20));         // 64 MB  scores f32 / P bf16
    bf16_t* P   = (bf16_t*)Sc;                           // in-place bf16, ld = 2*S
    bf16_t* AO  = Qb;                                    // attn output reuses Q

    // 1) LayerNorm -> bf16
    ln_bf16_kernel<<<rows, 256, 0, stream>>>(x, gamma, beta, xn, D);

    // 2) Weight casts (D*D = 1M elems each)
    const int castGrid = (D * D) / (256 * 4);
    cast_bf16_kernel<<<castGrid, 256, 0, stream>>>(Wq, Wqb, D * D);
    cast_bf16_kernel<<<castGrid, 256, 0, stream>>>(Wk, Wkb, D * D);
    cast_bf16_kernel<<<castGrid, 256, 0, stream>>>(Wv, Wvb, D * D);
    cast_bf16_kernel<<<castGrid, 256, 0, stream>>>(Wo, Wob, D * D);

    dim3 blk(256);
    // 3) Q = xn @ Wq^T + bq   (Bt := Wq as stored)
    gemm_bf16_nt<1, true><<<dim3(D / TILE_N, rows / TILE_M, 1), blk, 0, stream>>>(
        xn, Wqb, Qb, bq, rows, D, D, D, D, D, 0, 0, 0, 1.0f);
    // 4) K = xn @ Wk^T + bk
    gemm_bf16_nt<1, true><<<dim3(D / TILE_N, rows / TILE_M, 1), blk, 0, stream>>>(
        xn, Wkb, Kb, bk, rows, D, D, D, D, D, 0, 0, 0, 1.0f);
    // 5) V^T[e][b*S+s] = (xn @ Wv^T + bv)^T   (transposed store, ldc = rows)
    gemm_bf16_nt<2, true><<<dim3(D / TILE_N, rows / TILE_M, 1), blk, 0, stream>>>(
        xn, Wvb, Vt, bv, rows, D, D, D, D, rows, 0, 0, 0, 1.0f);
    // 6) scores = (Q @ K^T) / sqrt(D), batched over B, f32
    gemm_bf16_nt<0, false><<<dim3(S / TILE_N, S / TILE_M, B), blk, 0, stream>>>(
        Qb, Kb, Sc, nullptr, S, S, D, D, D, S,
        (long)S * D, (long)S * D, (long)S * S, inv_sqrt_d);
    // 7) softmax rows -> bf16 P in place
    softmax_inplace_kernel<<<rows, 256, 0, stream>>>(Sc, S);
    // 8) attn_out = P @ V  (A = P ld 2S, Bt = V^T slice, ldb = rows)
    gemm_bf16_nt<1, false><<<dim3(D / TILE_N, S / TILE_M, B), blk, 0, stream>>>(
        P, Vt, AO, nullptr, S, D, S, 2 * S, rows, D,
        (long)S * 2 * S, (long)S, (long)S * D, 1.0f);
    // 9) out = attn_out @ Wo^T + bo  (fp32 final)
    gemm_bf16_nt<0, true><<<dim3(D / TILE_N, rows / TILE_M, 1), blk, 0, stream>>>(
        AO, Wob, out, bo, rows, D, D, D, D, D, 0, 0, 0, 1.0f);
}
